// MPLMDecoderLayer_34059090657742
// MI455X (gfx1250) — matrix operand order repaired and checked by reading, not compiled, not run on hardware
//
#include <hip/hip_runtime.h>
#include <stdint.h>

// ---------------------------------------------------------------------------
// MI455X (gfx1250) decoder layer, round 5.
//   - all GEMMs + attention GEMMs on v_wmma_f32_16x16x32_bf16 (fp32 accum)
//   - GEMM LDS tiles staged by the Tensor Data Mover (tensor_load_to_lds,
//     TENSORcnt), one descriptor per 128x32 tile, HW padding -> pitch 40
//   - attention K/V chunks double-buffered via global_load_async_to_lds_b128
//     (ASYNCcnt); transposed score layout -> scalar per-lane softmax stats
// ---------------------------------------------------------------------------

typedef __bf16 bf16_t;
typedef __attribute__((ext_vector_type(16))) __bf16 v16bf;
typedef __attribute__((ext_vector_type(8)))  __bf16 v8bf;
typedef __attribute__((ext_vector_type(8)))  float  v8f;
typedef __attribute__((ext_vector_type(4)))  unsigned int v4u;
typedef __attribute__((ext_vector_type(4)))  int v4i;
typedef __attribute__((ext_vector_type(8)))  int v8i;

constexpr int B_   = 2;
constexpr int S_   = 2048;
constexpr int H_   = 1024;
constexpr int NH_  = 16;
constexpr int NKV_ = 4;
constexpr int HD_  = 64;
constexpr int FF_  = 4096;
constexpr int M_   = B_ * S_;   // 4096 tokens

__device__ __forceinline__ v8f wmma_bf16(v16bf a, v16bf b, v8f c) {
  return __builtin_amdgcn_wmma_f32_16x16x32_bf16(
      false, a, false, b, (short)0, c, false, false);
}

__device__ __forceinline__ v16bf cat8(v8bf lo, v8bf hi) {
  return __builtin_shufflevector(lo, hi, 0, 1, 2, 3, 4, 5, 6, 7,
                                         8, 9, 10, 11, 12, 13, 14, 15);
}

// A-operand fragment (16 rows x K=32), row-major source, leading dim ld.
__device__ __forceinline__ v16bf frag_row_a(const bf16_t* base, int ld) {
  const int lane = threadIdx.x & 31;
  const bf16_t* p = base + (lane & 15) * ld + ((lane >> 4) << 3);
  return cat8(*(const v8bf*)p, *(const v8bf*)(p + 16));
}

// B-operand fragment (K=32 x 16 cols), columns stored AS ROWS, leading dim ld.
__device__ __forceinline__ v16bf frag_row_b(const bf16_t* base, int ld) {
  const int lane = threadIdx.x & 31;
  const bf16_t* p = base + (lane & 15) * ld + ((lane >> 4) << 3);
  return cat8(*(const v8bf*)p, *(const v8bf*)(p + 16));
}

// ---- CDNA5 async global->LDS copy (ASYNCcnt path) --------------------------
__device__ __forceinline__ void async_b128(void* lds, const void* gptr) {
  const uint32_t l = (uint32_t)(uintptr_t)lds;   // LDS addr = generic[31:0]
  asm volatile("global_load_async_to_lds_b128 %0, %1, off"
               :: "v"(l), "v"(gptr) : "memory");
}
__device__ __forceinline__ void async_wait() {
  asm volatile("s_wait_asynccnt 0x0" ::: "memory");
}

// ---- CDNA5 Tensor Data Mover: 2D tile load, D# per ISA 8.3-8.5 -------------
// tile_d0 contiguous elements per row, tile_d1 rows, row stride (elements).
// pad_int_f/pad_amt_f are the encoded D# pad fields (pad after 2^(f+1) dwords,
// amount f+1 dwords), reproducing the padded LDS pitch in hardware.
__device__ __forceinline__ void tdm_load_2d(void* lds, const void* gsrc,
                                            uint32_t tile_d0, uint32_t tile_d1,
                                            uint32_t stride_e,
                                            uint32_t pad_int_f, uint32_t pad_amt_f) {
  const uint64_t ga = (uint64_t)(uintptr_t)gsrc;
  v4u g0;
  g0.x = 1u;                                             // count=1, user D#
  g0.y = (uint32_t)(uintptr_t)lds;                       // lds_addr
  g0.z = (uint32_t)ga;                                   // global_addr[31:0]
  g0.w = (uint32_t)((ga >> 32) & 0x01FFFFFFu) | (2u << 30);  // [56:32], type=2
  const uint32_t td0 = stride_e;        // tensor_dim0 (no OOB for our tiles)
  const uint32_t td1 = 1u << 20;        // tensor_dim1: large
  v8i g1;
  g1[0] = (int)((1u << 16) |            // data_size = 1 -> 2-byte elements
                (1u << 20) |            // pad_enable
                (pad_int_f << 22) | (pad_amt_f << 25));
  g1[1] = (int)((td0 & 0xFFFFu) << 16);                  // tensor_dim0 lo
  g1[2] = (int)((td0 >> 16) | ((td1 & 0xFFFFu) << 16));  // dim0 hi, dim1 lo
  g1[3] = (int)((td1 >> 16) | (tile_d0 << 16));          // dim1 hi, tile_dim0
  g1[4] = (int)tile_d1;                                  // tile_dim1, tile_dim2=0
  g1[5] = (int)stride_e;                                 // tensor_dim0_stride lo
  g1[6] = 0;                                             // stride hi, dim1_stride
  g1[7] = 0;
  const v4i z4 = {0, 0, 0, 0};
  const v8i z8 = {0, 0, 0, 0, 0, 0, 0, 0};
  __builtin_amdgcn_tensor_load_to_lds(g0, g1, z4, z4, z8, 0);
}
__device__ __forceinline__ void tdm_wait() {
  __builtin_amdgcn_s_wait_tensorcnt(0);
}

// ---------------------------------------------------------------------------
// fp32 -> bf16 transposing weight conversion: w[K][N] -> o[N][K]
// ---------------------------------------------------------------------------
__global__ void k_cvt_t(const float* __restrict__ w, bf16_t* __restrict__ o,
                        int K, int N) {
  const int idx = blockIdx.x * blockDim.x + threadIdx.x;
  if (idx >= K * N) return;
  const int k = idx % K;
  const int n = idx / K;
  o[idx] = (bf16_t)w[(size_t)k * N + n];
}

// ---------------------------------------------------------------------------
// RMSNorm (one row of H=1024 per block, fp32 stats, bf16 output)
// ---------------------------------------------------------------------------
__global__ void k_rmsnorm(const float* __restrict__ x, const float* __restrict__ w,
                          bf16_t* __restrict__ y) {
  __shared__ float red[256];
  const int row = blockIdx.x;
  const int tid = threadIdx.x;
  const float* xr = x + (size_t)row * H_;
  float s = 0.f;
  for (int i = tid; i < H_; i += 256) { float v = xr[i]; s += v * v; }
  red[tid] = s;
  __syncthreads();
  for (int o = 128; o > 0; o >>= 1) {
    if (tid < o) red[tid] += red[tid + o];
    __syncthreads();
  }
  const float r = rsqrtf(red[0] / (float)H_ + 1e-6f);
  for (int i = tid; i < H_; i += 256)
    y[(size_t)row * H_ + i] = (bf16_t)(xr[i] * r * w[i]);
}

// ---------------------------------------------------------------------------
// GEMM  C[M,N] = X[M,K] * W,  W pre-transposed: Wt[N][K].
// A = Wt tile (16n x 32k), B = X^T tile (32k x 16m), D(n in VGPRs, m in lanes).
// 256 threads / 8 waves, 128x128 block tile, K-step 32.
// Staging: TDM tensor_load_to_lds, one descriptor per tile, double-buffered.
// LDS pitch 40 elems produced by D# padding (4 dwords after every 16 dwords).
// ---------------------------------------------------------------------------
template <bool HAS_RES>
__global__ void k_gemm_bf16(const bf16_t* __restrict__ Wt,  // [N][K]
                            const bf16_t* __restrict__ X,   // [M][K]
                            float* __restrict__ C,          // [M][N]
                            const float* __restrict__ resid,
                            int M, int N, int K) {
  __shared__ bf16_t Wls[2][128][40];
  __shared__ bf16_t Xls[2][128][40];

  const int tid  = threadIdx.x;
  const int wv   = tid >> 5;
  const int lane = tid & 31;
  const int wn = wv >> 1;          // 0..3 : 32-wide n slice
  const int wm = wv & 1;           // 0..1 : 64-wide m slice
  const int n0 = blockIdx.x * 128;
  const int m0 = blockIdx.y * 128;
  const int sr = tid >> 1;         // used for prefetch row

  auto stage = [&](int buf, int k0) {
    if (wv == 0) {  // TDM descriptors are per-wave; one wave programs the DMA
      tdm_load_2d(&Wls[buf][0][0], Wt + (size_t)n0 * K + k0,
                  32, 128, (uint32_t)K, 3, 3);
      tdm_load_2d(&Xls[buf][0][0], X + (size_t)m0 * K + k0,
                  32, 128, (uint32_t)K, 3, 3);
    }
  };

  const v8f vzero = {0.f, 0.f, 0.f, 0.f, 0.f, 0.f, 0.f, 0.f};
  v8f acc[4][2];
#pragma unroll
  for (int i = 0; i < 4; ++i)
#pragma unroll
    for (int j = 0; j < 2; ++j) acc[i][j] = vzero;

  stage(0, 0);
  tdm_wait();
  __syncthreads();

  for (int k0 = 0; k0 < K; k0 += 32) {
    const int cur = (k0 >> 5) & 1;
    if (k0 + 32 < K) {
      stage(cur ^ 1, k0 + 32);                      // overlap DMA with WMMA
      __builtin_prefetch(Wt + (size_t)(n0 + sr) * K + k0 + 64, 0, 1);
      __builtin_prefetch(X  + (size_t)(m0 + sr) * K + k0 + 64, 0, 1);
    }

    v16bf af[2], bx[4];
#pragma unroll
    for (int j = 0; j < 2; ++j)
      af[j] = frag_row_a(&Wls[cur][wn * 32 + j * 16][0], 40);
#pragma unroll
    for (int i = 0; i < 4; ++i)
      bx[i] = frag_row_b(&Xls[cur][wm * 64 + i * 16][0], 40);
#pragma unroll
    for (int i = 0; i < 4; ++i)
#pragma unroll
      for (int j = 0; j < 2; ++j)
        acc[i][j] = wmma_bf16(af[j], bx[i], acc[i][j]);

    tdm_wait();
    __syncthreads();
  }

  const int hh = lane >> 4;
  const int ln = lane & 15;
#pragma unroll
  for (int i = 0; i < 4; ++i) {
    const int m = m0 + wm * 64 + i * 16 + ln;
#pragma unroll
    for (int j = 0; j < 2; ++j) {
      const int nb = n0 + wn * 32 + j * 16 + 8 * hh;
      const size_t idx = (size_t)m * N + nb;
#pragma unroll
      for (int r = 0; r < 8; ++r) {      // 8 consecutive floats -> b128 stores
        float v = acc[i][j][r];
        if (HAS_RES) v += resid[idx + r];
        C[idx + r] = v;
      }
    }
  }
}

// ---------------------------------------------------------------------------
// RoPE + repack [B*S, nh*HD] fp32 -> [B, nh, S, HD] bf16
// ---------------------------------------------------------------------------
__global__ void k_rope_pack(const float* __restrict__ in, bf16_t* __restrict__ out,
                            int nheads) {
  const int idx = blockIdx.x * blockDim.x + threadIdx.x;
  if (idx >= B_ * S_ * HD_ * nheads) return;
  const int d = idx & (HD_ - 1);
  const int s = (idx / HD_) % S_;
  const int h = (idx / (HD_ * S_)) % nheads;
  const int b = idx / (HD_ * S_ * nheads);
  const size_t src = ((size_t)(b * S_ + s)) * (nheads * HD_) + h * HD_ + d;
  const float inv = __powf(10000.f, -(float)(d & 31) / 32.f);
  float sn, cs;
  __sincosf((float)s * inv, &sn, &cs);
  const float other = (d < 32) ? -in[src + 32] : in[src - 32];
  out[idx] = (bf16_t)(in[src] * cs + other * sn);
}

// V pack, TRANSPOSED per head: [B*S, NKV*HD] fp32 -> [B, NKV, HD, S] bf16
__global__ void k_pack_vT(const float* __restrict__ in, bf16_t* __restrict__ out) {
  const int idx = blockIdx.x * blockDim.x + threadIdx.x;
  if (idx >= B_ * NKV_ * HD_ * S_) return;
  const int s = idx % S_;
  const int d = (idx / S_) % HD_;
  const int h = (idx / (S_ * HD_)) % NKV_;
  const int b = idx / (S_ * HD_ * NKV_);
  out[idx] = (bf16_t)in[((size_t)(b * S_ + s)) * (NKV_ * HD_) + h * HD_ + d];
}

// ---------------------------------------------------------------------------
// GQA flash attention, transposed score layout.  Grid: (S/64, NH, B), 128 thr.
// S^T = K x Q^T  -> D(key in VGPRs, q in lanes): scalar per-lane softmax stats
// O^T = V^T x P^T -> D(hd in VGPRs, q in lanes): 16B packed output stores
// K/V chunks double-buffered via async-to-LDS (ASYNCcnt).
// ---------------------------------------------------------------------------
__global__ void k_attention(const bf16_t* __restrict__ Q,    // [B,NH,S,HD]
                            const bf16_t* __restrict__ Kt,   // [B,NKV,S,HD]
                            const bf16_t* __restrict__ Vt,   // [B,NKV,HD,S]
                            const float* __restrict__ mask,  // [B,1,S,S]
                            bf16_t* __restrict__ Out) {      // [B*S, NH*HD]
  __shared__ bf16_t Qs[64][72];
  __shared__ bf16_t Ks[2][64][72];     // keys x hd
  __shared__ bf16_t Vts[2][64][72];    // hd x keys
  __shared__ bf16_t Ps[4][16][72];     // per-wave P rows (q x keys)

  const int qc = blockIdx.x, h = blockIdx.y, b = blockIdx.z;
  const int kvh = h / (NH_ / NKV_);
  const int tid  = threadIdx.x;
  const int wave = tid >> 5;
  const int lane = tid & 31;
  const int hh = lane >> 4;
  const int ln = lane & 15;
  const int q0 = qc * 64;
  const int sr = tid >> 1;             // staging row 0..63
  const int sc = (tid & 1) * 32;       // staging elem offset 0/32

  auto stage_kv = [&](int chunk, int buf) {
    const bf16_t* ksrc =
        Kt + (((size_t)b * NKV_ + kvh) * S_ + chunk * 64 + sr) * HD_ + sc;
    const bf16_t* vsrc =
        Vt + (((size_t)b * NKV_ + kvh) * HD_ + sr) * S_ + chunk * 64 + sc;
#pragma unroll
    for (int s = 0; s < 4; ++s) {
      async_b128(&Ks[buf][sr][sc + 8 * s],  ksrc + 8 * s);
      async_b128(&Vts[buf][sr][sc + 8 * s], vsrc + 8 * s);
    }
  };

  {  // stage Q once + first K/V chunk
    const bf16_t* qsrc =
        Q + (((size_t)b * NH_ + h) * S_ + q0 + sr) * HD_ + sc;
#pragma unroll
    for (int s = 0; s < 4; ++s) async_b128(&Qs[sr][sc + 8 * s], qsrc + 8 * s);
    stage_kv(0, 0);
  }
  async_wait();
  __syncthreads();

  // Q^T B-fragments for this wave (lane = q)
  const v16bf qb0 = frag_row_b(&Qs[wave * 16][0],  72);
  const v16bf qb1 = frag_row_b(&Qs[wave * 16][32], 72);

  const v8f vzero = {0.f, 0.f, 0.f, 0.f, 0.f, 0.f, 0.f, 0.f};
  v8f oacc[4];
#pragma unroll
  for (int t = 0; t < 4; ++t) oacc[t] = vzero;
  float mrow = -1e30f, lrow = 0.f;        // scalar per-lane stats (one q/lane)
  const float* mbase = mask + ((size_t)b * S_ + q0 + wave * 16 + ln) * S_;

  for (int kc = 0; kc < S_ / 64; ++kc) {
    const int cur = kc & 1;
    if (kc + 1 < S_ / 64) stage_kv(kc + 1, cur ^ 1);   // overlap DMA

    // hoist this chunk's mask values (8 consecutive floats per lane/tile)
    v8f mva[4];
#pragma unroll
    for (int kt = 0; kt < 4; ++kt)
      mva[kt] = *(const v8f*)(mbase + kc * 64 + kt * 16 + 8 * hh);

    // ---- scores: S^T tiles (key = kt*16 + r + 8*hh, q = lane) ----
    v8f sacc[4];
#pragma unroll
    for (int kt = 0; kt < 4; ++kt) {
      v8f s = vzero;
      s = wmma_bf16(frag_row_a(&Ks[cur][kt * 16][0],  72), qb0, s);
      s = wmma_bf16(frag_row_a(&Ks[cur][kt * 16][32], 72), qb1, s);
#pragma unroll
      for (int r = 0; r < 8; ++r) s[r] = s[r] * 0.125f + mva[kt][r];
      sacc[kt] = s;
    }

    // ---- online softmax (in-lane reduce + one shfl_xor(16)) ----
    float rm = -1e30f;
#pragma unroll
    for (int kt = 0; kt < 4; ++kt)
#pragma unroll
      for (int r = 0; r < 8; ++r) rm = fmaxf(rm, sacc[kt][r]);
    rm = fmaxf(rm, __shfl_xor(rm, 16, 32));
    const float mn = fmaxf(mrow, rm);
    const float alpha = __expf(mrow - mn);
    mrow = mn;
    float rs = 0.f;
#pragma unroll
    for (int kt = 0; kt < 4; ++kt)
#pragma unroll
      for (int r = 0; r < 8; ++r) {
        const float p = __expf(sacc[kt][r] - mn);
        sacc[kt][r] = p;
        rs += p;
      }
    rs += __shfl_xor(rs, 16, 32);
    lrow = lrow * alpha + rs;
#pragma unroll
    for (int t = 0; t < 4; ++t)
#pragma unroll
      for (int r = 0; r < 8; ++r) oacc[t][r] *= alpha;

    // ---- P rows to LDS: 16B packed stores (q = lane row) ----
#pragma unroll
    for (int kt = 0; kt < 4; ++kt) {
      v8bf pk;
#pragma unroll
      for (int r = 0; r < 8; ++r) pk[r] = (bf16_t)sacc[kt][r];
      *(v8bf*)&Ps[wave][ln][kt * 16 + 8 * hh] = pk;
    }

    // ---- O^T += V^T x P^T ----
    const v16bf pb0 = frag_row_b(&Ps[wave][0][0],  72);
    const v16bf pb1 = frag_row_b(&Ps[wave][0][32], 72);
#pragma unroll
    for (int ot = 0; ot < 4; ++ot) {
      oacc[ot] = wmma_bf16(frag_row_a(&Vts[cur][ot * 16][0],  72), pb0, oacc[ot]);
      oacc[ot] = wmma_bf16(frag_row_a(&Vts[cur][ot * 16][32], 72), pb1, oacc[ot]);
    }

    async_wait();
    __syncthreads();
  }

  // ---- normalize, pack, 16B stores ----
  const float inv = 1.f / lrow;
  const size_t row = (size_t)b * S_ + q0 + wave * 16 + ln;
#pragma unroll
  for (int ot = 0; ot < 4; ++ot) {
    v8bf ob;
#pragma unroll
    for (int r = 0; r < 8; ++r) ob[r] = (bf16_t)(oacc[ot][r] * inv);
    *(v8bf*)&Out[row * (NH_ * HD_) + h * HD_ + ot * 16 + 8 * hh] = ob;
  }
}

// ---------------------------------------------------------------------------
// SwiGLU elementwise: act = silu(gate) * up  (fp32 in, bf16 out)
// ---------------------------------------------------------------------------
__global__ void k_silu_mul(const float* __restrict__ g, const float* __restrict__ u,
                           bf16_t* __restrict__ out, int n) {
  const int i = blockIdx.x * blockDim.x + threadIdx.x;
  if (i < n) {
    const float gv = g[i];
    out[i] = (bf16_t)(gv / (1.f + __expf(-gv)) * u[i]);
  }
}

// ---------------------------------------------------------------------------
// Host orchestration
// ---------------------------------------------------------------------------
extern "C" void kernel_launch(void* const* d_in, const int* in_sizes, int n_in,
                              void* d_out, int out_size, void* d_ws, size_t ws_size,
                              hipStream_t stream) {
  (void)in_sizes; (void)n_in; (void)out_size; (void)ws_size;

  const float* hidden = (const float*)d_in[0];
  const float* kvs    = (const float*)d_in[1];
  const float* amask  = (const float*)d_in[2];
  const float* ln1w   = (const float*)d_in[3];
  const float* ln2w   = (const float*)d_in[4];
  const float* Wq     = (const float*)d_in[5];
  const float* Wk     = (const float*)d_in[6];
  const float* Wv     = (const float*)d_in[7];
  const float* Wo     = (const float*)d_in[8];
  const float* Wg     = (const float*)d_in[9];
  const float* Wu     = (const float*)d_in[10];
  const float* Wd     = (const float*)d_in[11];
  float* out = (float*)d_out;

  uint8_t* ws = (uint8_t*)d_ws;
  size_t off = 0;
  auto alloc = [&](size_t bytes) -> void* {
    void* p = ws + off;
    off = (off + bytes + 255) & ~(size_t)255;
    return p;
  };

  bf16_t* wq_b = (bf16_t*)alloc((size_t)H_ * NH_ * HD_ * 2);   // [NH*HD][H]
  bf16_t* wk_b = (bf16_t*)alloc((size_t)H_ * NKV_ * HD_ * 2);  // [NKV*HD][H]
  bf16_t* wv_b = (bf16_t*)alloc((size_t)H_ * NKV_ * HD_ * 2);
  bf16_t* wo_b = (bf16_t*)alloc((size_t)NH_ * HD_ * H_ * 2);   // [H][NH*HD]
  bf16_t* wg_b = (bf16_t*)alloc((size_t)H_ * FF_ * 2);         // [FF][H]
  bf16_t* wu_b = (bf16_t*)alloc((size_t)H_ * FF_ * 2);
  bf16_t* wd_b = (bf16_t*)alloc((size_t)FF_ * H_ * 2);         // [H][FF]
  bf16_t* xn1  = (bf16_t*)alloc((size_t)M_ * H_ * 2);
  bf16_t* kvn  = (bf16_t*)alloc((size_t)M_ * H_ * 2);
  float*  qf   = (float*)alloc((size_t)M_ * NH_ * HD_ * 4);
  float*  kf   = (float*)alloc((size_t)M_ * NKV_ * HD_ * 4);
  float*  vf   = (float*)alloc((size_t)M_ * NKV_ * HD_ * 4);
  bf16_t* qb   = (bf16_t*)alloc((size_t)M_ * NH_ * HD_ * 2);
  bf16_t* kb   = (bf16_t*)alloc((size_t)M_ * NKV_ * HD_ * 2);
  bf16_t* vb   = (bf16_t*)alloc((size_t)M_ * NKV_ * HD_ * 2);
  bf16_t* atnb = (bf16_t*)alloc((size_t)M_ * NH_ * HD_ * 2);
  float*  hid  = (float*)alloc((size_t)M_ * H_ * 4);
  bf16_t* x2   = (bf16_t*)alloc((size_t)M_ * H_ * 2);
  float*  gf   = (float*)alloc((size_t)M_ * FF_ * 4);
  float*  uf   = (float*)alloc((size_t)M_ * FF_ * 4);
  bf16_t* actb = (bf16_t*)alloc((size_t)M_ * FF_ * 2);

  auto cvtT = [&](const float* src, bf16_t* dst, int K, int N) {
    k_cvt_t<<<(K * N + 255) / 256, 256, 0, stream>>>(src, dst, K, N);
  };

  // weights -> bf16, transposed to [N][K]
  cvtT(Wq, wq_b, H_, NH_ * HD_);
  cvtT(Wk, wk_b, H_, NKV_ * HD_);
  cvtT(Wv, wv_b, H_, NKV_ * HD_);
  cvtT(Wo, wo_b, NH_ * HD_, H_);
  cvtT(Wg, wg_b, H_, FF_);
  cvtT(Wu, wu_b, H_, FF_);
  cvtT(Wd, wd_b, FF_, H_);

  // norms
  k_rmsnorm<<<M_, 256, 0, stream>>>(hidden, ln1w, xn1);
  k_rmsnorm<<<M_, 256, 0, stream>>>(kvs,    ln1w, kvn);

  // QKV projections
  k_gemm_bf16<false><<<dim3((NH_ * HD_) / 128, M_ / 128), 256, 0, stream>>>(
      wq_b, xn1, qf, nullptr, M_, NH_ * HD_, H_);
  k_gemm_bf16<false><<<dim3((NKV_ * HD_) / 128, M_ / 128), 256, 0, stream>>>(
      wk_b, kvn, kf, nullptr, M_, NKV_ * HD_, H_);
  k_gemm_bf16<false><<<dim3((NKV_ * HD_) / 128, M_ / 128), 256, 0, stream>>>(
      wv_b, kvn, vf, nullptr, M_, NKV_ * HD_, H_);

  // RoPE + layouts
  k_rope_pack<<<(B_ * S_ * NH_ * HD_) / 256, 256, 0, stream>>>(qf, qb, NH_);
  k_rope_pack<<<(B_ * S_ * NKV_ * HD_) / 256, 256, 0, stream>>>(kf, kb, NKV_);
  k_pack_vT<<<(B_ * S_ * NKV_ * HD_) / 256, 256, 0, stream>>>(vf, vb);

  // GQA flash attention
  k_attention<<<dim3(S_ / 64, NH_, B_), 128, 0, stream>>>(qb, kb, vb, amask, atnb);

  // O projection + residual
  k_gemm_bf16<true><<<dim3(H_ / 128, M_ / 128), 256, 0, stream>>>(
      wo_b, atnb, hid, hidden, M_, H_, NH_ * HD_);

  // MLP
  k_rmsnorm<<<M_, 256, 0, stream>>>(hid, ln2w, x2);
  k_gemm_bf16<false><<<dim3(FF_ / 128, M_ / 128), 256, 0, stream>>>(
      wg_b, x2, gf, nullptr, M_, FF_, H_);
  k_gemm_bf16<false><<<dim3(FF_ / 128, M_ / 128), 256, 0, stream>>>(
      wu_b, x2, uf, nullptr, M_, FF_, H_);
  k_silu_mul<<<(M_ * FF_) / 256, 256, 0, stream>>>(gf, uf, actb, M_ * FF_);
  k_gemm_bf16<true><<<dim3(H_ / 128, M_ / 128), 256, 0, stream>>>(
      wd_b, actb, out, hid, M_, H_, FF_);
}
